// GraphAttentionLayer_20633022890601
// MI455X (gfx1250) — matrix operand order repaired and checked by reading, not compiled, not run on hardware
//
#include <hip/hip_runtime.h>

#define Bb   32
#define Nn   512
#define FIN  256
#define Hh   8
#define Ff   64
#define HF   512          // Hh*Ff
#define ROWS (Bb*Nn)      // 16384
#define NEGINF (-9.0e15f)
#define ALPHA  0.2f

typedef __attribute__((ext_vector_type(16))) __bf16 v16bf;
typedef __attribute__((ext_vector_type(8)))  float  v8f;
typedef __attribute__((ext_vector_type(4)))  unsigned v4u;
typedef __attribute__((ext_vector_type(8)))  unsigned v8u;

union BF16x16 { v16bf v; __bf16 e[16]; };

__device__ __forceinline__ v8f wmma_bf16(v16bf a, v16bf b, v8f c) {
  // D = A(16x32 bf16) * B(32x16 bf16) + C(16x16 f32)
  return __builtin_amdgcn_wmma_f32_16x16x32_bf16(false, a, false, b, (short)0, c, false, false);
}

#define RFL(x) ((unsigned)__builtin_amdgcn_readfirstlane((int)(x)))

// ---------------------------------------------------------------------------
// Tensor Data Mover: 2D tile (global f32 -> LDS), normal mode, optional LDS
// padding. D# layout per CDNA5 ISA.
// Issued once per calling wave (EXEC ignored); completion via TENSORcnt.
// ---------------------------------------------------------------------------
__device__ __forceinline__ void tdm_load_2d_f32(unsigned lds_off, const void* gptr,
                                                unsigned tensor_d0, unsigned tensor_d1,
                                                unsigned tile_d0, unsigned tile_d1,
                                                unsigned stride_d0,
                                                unsigned pad_en, unsigned pad_ivl,
                                                unsigned pad_amt) {
  const unsigned long long ga = (unsigned long long)gptr;
  v4u g0;
  g0[0] = RFL(1u);                                        // count=1, user desc
  g0[1] = RFL(lds_off);                                   // lds_addr (bytes)
  g0[2] = RFL((unsigned)(ga & 0xFFFFFFFFu));              // global_addr[31:0]
  g0[3] = RFL(((unsigned)((ga >> 32) & 0x01FFFFFFu)) | (2u << 30)); // ga[56:32]|type=2
  v8u g1;
  g1[0] = RFL((2u << 16) | (pad_en << 20) | (pad_ivl << 22) | (pad_amt << 25)); // 4B elems
  g1[1] = RFL((tensor_d0 & 0xFFFFu) << 16);               // tensor_dim0[15:0] @48
  g1[2] = RFL((tensor_d0 >> 16) | ((tensor_d1 & 0xFFFFu) << 16)); // d0 hi | d1 lo
  g1[3] = RFL((tensor_d1 >> 16) | (tile_d0 << 16));       // d1 hi | tile_dim0 @112
  g1[4] = RFL(tile_d1);                                   // tile_dim1 @128, tile_dim2=0
  g1[5] = RFL(stride_d0);                                 // tensor_dim0_stride[31:0]
  g1[6] = RFL(0u);
  g1[7] = RFL(0u);
  asm volatile("tensor_load_to_lds %0, %1" :: "s"(g0), "s"(g1) : "memory");
}

// ---------------------------------------------------------------------------
// Kernel 1: h = x @ W  (M=16384, K=256, N=512), fp32 in -> bf16 out (+ bf16
// transposed copy per batch for GEMM2's B fragments). Block 256 = 8 waves,
// block tile 128(M) x 64(N), each wave owns 16(M) x 64(N) = 4 WMMA C tiles.
// ---------------------------------------------------------------------------
__global__ __launch_bounds__(256) void k_gemm_xw(const float* __restrict__ x,
                                                 const float* __restrict__ W,
                                                 __bf16* __restrict__ hbf,
                                                 __bf16* __restrict__ hT) {
  __shared__ __bf16 lx[128][34];   // x tile, row-major [m][k]
  __shared__ __bf16 lw[64][34];    // W tile, transposed [n][k]
  const int tid  = threadIdx.x;
  const int lane = tid & 31;
  const int wave = tid >> 5;
  const int n0 = blockIdx.x * 64;
  const int m0 = blockIdx.y * 128;

  v8f acc[4] = {};

  for (int kc = 0; kc < FIN / 32; ++kc) {
    const int k0 = kc * 32;
    __syncthreads();
    // issue all global loads first so they pipeline, then convert+store to LDS
    float4 vx[4], vw[2];
    #pragma unroll
    for (int t = 0; t < 4; ++t) {
      int f4 = tid + t * 256;
      int r = f4 >> 3, kk = (f4 & 7) << 2;
      vx[t] = *(const float4*)(x + (size_t)(m0 + r) * FIN + k0 + kk);
    }
    #pragma unroll
    for (int t = 0; t < 2; ++t) {
      int f4 = tid + t * 256;
      int kr = f4 >> 4, nn = (f4 & 15) << 2;
      vw[t] = *(const float4*)(W + (size_t)(k0 + kr) * HF + n0 + nn);
    }
    #pragma unroll
    for (int t = 0; t < 4; ++t) {
      int f4 = tid + t * 256;
      int r = f4 >> 3, kk = (f4 & 7) << 2;
      lx[r][kk + 0] = (__bf16)vx[t].x; lx[r][kk + 1] = (__bf16)vx[t].y;
      lx[r][kk + 2] = (__bf16)vx[t].z; lx[r][kk + 3] = (__bf16)vx[t].w;
    }
    #pragma unroll
    for (int t = 0; t < 2; ++t) {
      int f4 = tid + t * 256;
      int kr = f4 >> 4, nn = (f4 & 15) << 2;
      lw[nn + 0][kr] = (__bf16)vw[t].x; lw[nn + 1][kr] = (__bf16)vw[t].y;
      lw[nn + 2][kr] = (__bf16)vw[t].z; lw[nn + 3][kr] = (__bf16)vw[t].w;
    }
    __syncthreads();

    // A fragment: 16x32 bf16 (contiguous: lanes0-15 K=0..15, lanes16-31 K=16..31)
    BF16x16 a;
    const int ml    = wave * 16 + (lane & 15);
    const int khalf = (lane >> 4) << 4;
    #pragma unroll
    for (int v = 0; v < 8; ++v) {
      int k = khalf + 2 * v;
      a.e[2 * v]     = lx[ml][k];
      a.e[2 * v + 1] = lx[ml][k + 1];
    }
    #pragma unroll
    for (int nt = 0; nt < 4; ++nt) {
      BF16x16 bf;
      const int nl = nt * 16 + (lane & 15);
      const int kb = (lane >> 4) << 4;
      #pragma unroll
      for (int v = 0; v < 8; ++v) {
        bf.e[2 * v]     = lw[nl][kb + 2 * v];
        bf.e[2 * v + 1] = lw[nl][kb + 2 * v + 1];
      }
      acc[nt] = wmma_bf16(a.v, bf.v, acc[nt]);
    }
  }

  // epilogue: C layout = VGPR r -> M = r + 8*(lane>=16), N = lane&15
  const int rbase = m0 + wave * 16 + ((lane >> 4) << 3);
  #pragma unroll
  for (int nt = 0; nt < 4; ++nt) {
    const int col = n0 + nt * 16 + (lane & 15);
    #pragma unroll
    for (int r = 0; r < 8; ++r) {
      const int row = rbase + r;
      __bf16 hv = (__bf16)acc[nt][r];
      hbf[(size_t)row * HF + col] = hv;
      const int b = row >> 9, i = row & (Nn - 1);
      hT[((size_t)b * HF + col) * Nn + i] = hv;
    }
  }
}

// ---------------------------------------------------------------------------
// Kernel 2: s_src[row,h] = h[row,h,:].a_src[h],  s_dst likewise (plus a
// per-batch transposed copy of s_dst so kernel 4 can TDM it into LDS).
// One thread per (row, head).
// ---------------------------------------------------------------------------
__global__ __launch_bounds__(256) void k_scores(const __bf16* __restrict__ hbf,
                                                const float* __restrict__ a,
                                                float* __restrict__ ssrc,
                                                float* __restrict__ sdst,
                                                float* __restrict__ sdstT) {
  const int t = blockIdx.x * 256 + threadIdx.x;   // t = row*Hh + head
  const int row = t >> 3, head = t & 7;
  const __bf16* hp  = hbf + (size_t)row * HF + head * Ff;
  const float*  asr = a + head * 2 * Ff;
  const float*  ads = asr + Ff;
  float s1 = 0.f, s2 = 0.f;
  #pragma unroll 8
  for (int f = 0; f < Ff; ++f) {
    const float hv = (float)hp[f];
    s1 += hv * asr[f];
    s2 += hv * ads[f];
  }
  ssrc[t] = s1;
  sdst[t] = s2;
  const int b = row >> 9, i = row & (Nn - 1);
  sdstT[((size_t)b * Hh + head) * Nn + i] = s2;
}

// ---------------------------------------------------------------------------
// Kernel 3: softmax pass 1 — row max m and sum l per (b,i,head).
// One block per row (b,i); adj row staged once in LDS; wave = head.
// ---------------------------------------------------------------------------
__global__ __launch_bounds__(256) void k_stats(const float* __restrict__ adj,
                                               const float* __restrict__ ssrc,
                                               const float* __restrict__ sdst,
                                               float* __restrict__ mst,
                                               float* __restrict__ lst) {
  __shared__ float ladj[Nn];
  const int row  = blockIdx.x;          // row = b*Nn + i
  const int b    = row >> 9;
  const int tid  = threadIdx.x;
  const int lane = tid & 31;
  const int head = tid >> 5;
  #pragma unroll
  for (int t = 0; t < 2; ++t)
    ladj[tid + t * 256] = adj[(size_t)row * Nn + tid + t * 256];
  __syncthreads();

  const float si = ssrc[row * Hh + head];
  float m = -3.0e38f, l = 0.f;
  for (int j = lane; j < Nn; j += 32) {
    float e = si + sdst[((size_t)(b << 9) + j) * Hh + head];
    e = (e > 0.f) ? e : ALPHA * e;
    e = (ladj[j] > 0.f) ? e : NEGINF;      // finite mask: all math stays finite
    if (e > m) { l = l * __expf(m - e) + 1.f; m = e; }
    else       { l += __expf(e - m); }
  }
  #pragma unroll
  for (int off = 16; off > 0; off >>= 1) {
    const float m2 = __shfl_xor(m, off, 32);
    const float l2 = __shfl_xor(l, off, 32);
    const float mn = fmaxf(m, m2);
    l = l * __expf(m - mn) + l2 * __expf(m2 - mn);
    m = mn;
  }
  if (lane == 0) { mst[row * Hh + head] = m; lst[row * Hh + head] = l; }
}

// ---------------------------------------------------------------------------
// Kernel 4: h_prime = softmax(e) @ h, fused (P never materialized).
// Block = (b, 16-row tile); 8 waves = 8 heads. The 16x512 adj tile and the
// 8x512 s_dst slice are staged into LDS by the Tensor Data Mover (waves 0/1
// issue one tensor_load_to_lds each; TDM inserts 2 pad DWORDs per 256 so the
// adj row footprint is 516 floats -> conflict-free row-indexed reads).
// Per 32-j chunk: build normalized P tile as bf16 A fragment (exp/l folded),
// B fragments from transposed bf16 h; 4 WMMAs -> 16x64 fp32 output per wave.
// ---------------------------------------------------------------------------
__global__ __launch_bounds__(256) void k_apply(const float* __restrict__ adj,
                                               const float* __restrict__ ssrc,
                                               const float* __restrict__ sdstT,
                                               const float* __restrict__ mst,
                                               const float* __restrict__ lst,
                                               const __bf16* __restrict__ hT,
                                               float* __restrict__ out) {
  __shared__ float ladj[16][516];   // TDM-padded: elem (r,j) at [r][j + 2*(j>>8)]
  __shared__ float lsd[Hh][Nn];
  const int b    = blockIdx.x;
  const int i0   = blockIdx.y * 16;
  const int tid  = threadIdx.x;
  const int lane = tid & 31;
  const int wave = tid >> 5;
  const int head = wave;

  if (wave == 0) {
    // 16 rows x 512 f32, global row stride 512; pad 2 DWORDs per 256 DWORDs
    tdm_load_2d_f32((unsigned)(unsigned long long)(void*)&ladj[0][0],
                    adj + ((size_t)(b * Nn) + i0) * Nn,
                    /*tensor_d0=*/512, /*tensor_d1=*/16,
                    /*tile_d0=*/512,   /*tile_d1=*/16,
                    /*stride_d0=*/512, /*pad_en=*/1, /*pad_ivl=*/7, /*pad_amt=*/1);
  } else if (wave == 1) {
    // contiguous 8*512 f32 slice of sdstT for batch b
    tdm_load_2d_f32((unsigned)(unsigned long long)(void*)&lsd[0][0],
                    sdstT + (size_t)b * Hh * Nn,
                    /*tensor_d0=*/4096, /*tensor_d1=*/1,
                    /*tile_d0=*/4096,   /*tile_d1=*/1,
                    /*stride_d0=*/4096, /*pad_en=*/0, /*pad_ivl=*/0, /*pad_amt=*/0);
  }
  __builtin_amdgcn_s_wait_tensorcnt((short)0);
  __syncthreads();

  const int   iloc = lane & 15;
  const int   row  = b * Nn + i0 + iloc;
  const float si   = ssrc[row * Hh + head];
  const float mi   = mst[row * Hh + head];
  const float il   = 1.0f / lst[row * Hh + head];
  const int   jhalf = (lane >> 4) << 4;
  const int   kboff = (lane >> 4) << 4;

  v8f acc[4] = {};
  for (int jc = 0; jc < Nn / 32; ++jc) {
    const int j0 = jc * 32;
    BF16x16 a;
    #pragma unroll
    for (int v = 0; v < 8; ++v) {
      const int kk = jhalf + 2 * v;
      #pragma unroll
      for (int e2 = 0; e2 < 2; ++e2) {
        const int j  = j0 + kk + e2;
        const int jp = j + ((j >> 8) << 1);          // TDM pad adjustment
        float ev = si + lsd[head][j];
        ev = (ev > 0.f) ? ev : ALPHA * ev;
        ev = (ladj[iloc][jp] > 0.f) ? ev : NEGINF;
        a.e[2 * v + e2] = (__bf16)(__expf(ev - mi) * il);   // P/l folded in
      }
    }
    #pragma unroll
    for (int nt = 0; nt < 4; ++nt) {
      BF16x16 bf;
      const int fg = head * Ff + nt * 16 + (lane & 15);
      const __bf16* hp = hT + ((size_t)b * HF + fg) * Nn + j0 + kboff;
      #pragma unroll
      for (int v = 0; v < 8; ++v) {
        bf.e[2 * v]     = hp[2 * v];
        bf.e[2 * v + 1] = hp[2 * v + 1];
      }
      acc[nt] = wmma_bf16(a.v, bf.v, acc[nt]);
    }
  }

  const int rbase = i0 + ((lane >> 4) << 3);
  #pragma unroll
  for (int nt = 0; nt < 4; ++nt) {
    const int col = head * Ff + nt * 16 + (lane & 15);
    #pragma unroll
    for (int r = 0; r < 8; ++r)
      out[((size_t)(b * Nn) + rbase + r) * HF + col] = acc[nt][r];
  }
}

// ---------------------------------------------------------------------------
extern "C" void kernel_launch(void* const* d_in, const int* in_sizes, int n_in,
                              void* d_out, int out_size, void* d_ws, size_t ws_size,
                              hipStream_t stream) {
  const float* x   = (const float*)d_in[0];   // (B,N,Fin)
  const float* adj = (const float*)d_in[1];   // (B,N,N)
  const float* W   = (const float*)d_in[2];   // (Fin, H*F)
  const float* a   = (const float*)d_in[3];   // (H, 2F)
  float* out = (float*)d_out;                 // (B,N,H*F)

  // workspace carve-up (~36.2 MB)
  char* w = (char*)d_ws;
  __bf16* hbf  = (__bf16*)w;                                   // 16.78 MB
  __bf16* hT   = (__bf16*)(w + (size_t)ROWS * HF * 2);         // 16.78 MB
  float* ssrc  = (float*)(w + (size_t)ROWS * HF * 4);          // 512 KB each
  float* sdst  = ssrc  + (size_t)ROWS * Hh;
  float* sdstT = sdst  + (size_t)ROWS * Hh;
  float* mst   = sdstT + (size_t)ROWS * Hh;
  float* lst   = mst   + (size_t)ROWS * Hh;

  k_gemm_xw<<<dim3(HF / 64, ROWS / 128), 256, 0, stream>>>(x, W, hbf, hT);
  k_scores <<<(ROWS * Hh) / 256, 256, 0, stream>>>(hbf, a, ssrc, sdst, sdstT);
  k_stats  <<<ROWS, 256, 0, stream>>>(adj, ssrc, sdst, mst, lst);
  k_apply  <<<dim3(Bb, Nn / 16), 256, 0, stream>>>(adj, ssrc, sdstT, mst, lst, hT, out);
}
